// TransformerBlock_27599459844512
// MI455X (gfx1250) — compile-verified, matrix operand order checked by reading, not run on hardware
//
#include <hip/hip_runtime.h>

// ---------------------------------------------------------------------------
// Transformer block for MI455X (gfx1250, wave32, WMMA 16x16x32 f16)
// B=2, T=2048, E=1024, H=16, hd=64
// ---------------------------------------------------------------------------

#define TB 2048
#define TE 1024
#define TH 16
#define THD 64
#define TM 4096          // B*T rows

typedef __attribute__((ext_vector_type(16))) _Float16 v16h;
typedef __attribute__((ext_vector_type(8)))  float    v8f;

union Frag16 { v16h v; unsigned int u[8]; };
union Acc8   { v8f  v; float f[8]; };
union U4H    { uint4 q; _Float16 h[8]; };

// --------------------------- async global->LDS -----------------------------
#if __has_builtin(__builtin_amdgcn_global_load_async_to_lds_b128)
#define HAS_ASYNC_LDS 1
#else
#define HAS_ASYNC_LDS 0
#endif

// The builtin expects pointers to 16-byte int vectors in addrspace(1)/(3).
typedef int gcc_v4i __attribute__((vector_size(16)));
typedef __attribute__((address_space(1))) gcc_v4i as1_v4i;
typedef __attribute__((address_space(3))) gcc_v4i as3_v4i;

__device__ __forceinline__ void copy_b128_to_lds(_Float16* lds_dst, const _Float16* gsrc) {
#if HAS_ASYNC_LDS
    __builtin_amdgcn_global_load_async_to_lds_b128(
        (as1_v4i*)(void*)gsrc, (as3_v4i*)(void*)lds_dst, 0, 0);
#else
    *(uint4*)lds_dst = *(const uint4*)gsrc;
#endif
}

__device__ __forceinline__ void async_copies_wait() {
#if HAS_ASYNC_LDS && __has_builtin(__builtin_amdgcn_s_wait_asynccnt)
    __builtin_amdgcn_s_wait_asynccnt(0);
#endif
}

// K-index mapping for 16-bit A/B WMMA fragments (CDNA5 ISA 7.12.2):
// VGPR r (r<4): K = 2r,2r+1 (+8 for lanes 16..31); r>=4: K = 16+2(r-4) (+8)
__device__ __forceinline__ int kmap(int r, int hi) {
    return ((r & 4) ? 16 : 0) + ((r & 3) * 2) + hi * 8;
}

__device__ __forceinline__ float warp16_max(float v) {
    v = fmaxf(v, __shfl_xor(v, 1, 32));
    v = fmaxf(v, __shfl_xor(v, 2, 32));
    v = fmaxf(v, __shfl_xor(v, 4, 32));
    v = fmaxf(v, __shfl_xor(v, 8, 32));
    return v;
}
__device__ __forceinline__ float warp16_sum(float v) {
    v += __shfl_xor(v, 1, 32);
    v += __shfl_xor(v, 2, 32);
    v += __shfl_xor(v, 4, 32);
    v += __shfl_xor(v, 8, 32);
    return v;
}

// ---------------------------------------------------------------------------
// LayerNorm (fp32 in -> f16 out), one block per row of E=1024
// ---------------------------------------------------------------------------
__global__ __launch_bounds__(256) void ln_kernel(const float* __restrict__ x,
                                                 const float* __restrict__ g,
                                                 const float* __restrict__ b,
                                                 _Float16* __restrict__ out) {
    const int E = TE;
    const int row = blockIdx.x;
    const int tid = threadIdx.x;
    const float* xr = x + (size_t)row * E;

    float vals[4];
    float lsum = 0.f, lsq = 0.f;
#pragma unroll
    for (int i = 0; i < 4; ++i) {
        vals[i] = xr[tid + i * 256];
        lsum += vals[i];
        lsq  += vals[i] * vals[i];
    }
    __shared__ float s1[256], s2[256];
    s1[tid] = lsum; s2[tid] = lsq;
    __syncthreads();
    for (int off = 128; off > 0; off >>= 1) {
        if (tid < off) { s1[tid] += s1[tid + off]; s2[tid] += s2[tid + off]; }
        __syncthreads();
    }
    const float mean = s1[0] * (1.0f / E);
    const float var  = s2[0] * (1.0f / E) - mean * mean;
    const float rstd = rsqrtf(var + 1e-5f);

    _Float16* orow = out + (size_t)row * E;
#pragma unroll
    for (int i = 0; i < 4; ++i) {
        const int idx = tid + i * 256;
        orow[idx] = (_Float16)((vals[i] - mean) * rstd * g[idx] + b[idx]);
    }
}

// ---------------------------------------------------------------------------
// fp32 -> f16 cast (weights)
// ---------------------------------------------------------------------------
__global__ __launch_bounds__(256) void cast_kernel(const float* __restrict__ src,
                                                   _Float16* __restrict__ dst, int n) {
    int i = blockIdx.x * blockDim.x + threadIdx.x;
    const int stride = gridDim.x * blockDim.x;
    for (; i < n; i += stride) dst[i] = (_Float16)src[i];
}

// ---------------------------------------------------------------------------
// Tiled WMMA GEMM: C[M,N] = A[M,K] @ W[K,N] (+bias, epilogue variants)
//   EPI 0: +bias, store f16
//   EPI 1: +bias, exact GELU, store f16
//   EPI 2: +bias + res*wscalar, store f32
// Block: 256 thr (8 waves), tile 128x256x32; wave = 64x64 (4x4 WMMA tiles)
// A tile copied with GLOBAL_LOAD_ASYNC_TO_LDS_B128 (ASYNCcnt path).
// ---------------------------------------------------------------------------
#define BM 128
#define BN 256
#define BK 32
#define LDT 40   // padded LDS K-stride (halves)

template <int EPI>
__global__ __launch_bounds__(256) void gemm_kernel(const _Float16* __restrict__ A,
                                                   const _Float16* __restrict__ W,
                                                   const float* __restrict__ bias,
                                                   _Float16* __restrict__ outh,
                                                   float* __restrict__ outf,
                                                   const float* __restrict__ res,
                                                   const float* __restrict__ wscalar,
                                                   int M, int N, int K) {
    __shared__ __align__(16) _Float16 As[BM * LDT];
    __shared__ __align__(16) _Float16 Bt[BN * LDT];   // stored [n][k]

    const int tid  = threadIdx.x;
    const int lane = tid & 31;
    const int wid  = tid >> 5;
    const int wm   = wid >> 2;          // 0..1
    const int wn   = wid & 3;           // 0..3
    const int lm   = lane & 15;
    const int hi   = lane >> 4;

    const int m0 = blockIdx.y * BM;
    const int n0 = blockIdx.x * BN;
    const int row0 = wm * 64;
    const int col0 = wn * 64;

    Acc8 acc[4][4];
#pragma unroll
    for (int mt = 0; mt < 4; ++mt)
#pragma unroll
        for (int nt = 0; nt < 4; ++nt)
#pragma unroll
            for (int r = 0; r < 8; ++r) acc[mt][nt].f[r] = 0.f;

    for (int kt = 0; kt < K; kt += BK) {
        // ---- A tile: 128 rows x 32 halves -> 512 b128 async copies, 2/thread
#pragma unroll
        for (int i = 0; i < 2; ++i) {
            const int idx = tid * 2 + i;
            const int row = idx >> 2;
            const int seg = idx & 3;
            copy_b128_to_lds(&As[row * LDT + seg * 8],
                             A + (size_t)(m0 + row) * K + kt + seg * 8);
        }
        // ---- B tile: 32 k-rows x 256 n -> scatter-transpose to Bt[n][k]
#pragma unroll
        for (int i = 0; i < 4; ++i) {
            const int idx  = tid * 4 + i;
            const int kr   = idx >> 5;          // 0..31
            const int nseg = idx & 31;          // 0..31
            U4H d;
            d.q = *(const uint4*)(W + (size_t)(kt + kr) * N + n0 + nseg * 8);
#pragma unroll
            for (int j = 0; j < 8; ++j) Bt[(nseg * 8 + j) * LDT + kr] = d.h[j];
            if (kt + BK < K)
                __builtin_prefetch(W + (size_t)(kt + BK + kr) * N + n0 + nseg * 8, 0, 3);
        }
        async_copies_wait();
        __syncthreads();

        Frag16 aF[4];
#pragma unroll
        for (int mt = 0; mt < 4; ++mt)
#pragma unroll
            for (int r = 0; r < 8; ++r)
                aF[mt].u[r] = *(const unsigned int*)&As[(row0 + mt * 16 + lm) * LDT + kmap(r, hi)];

#pragma unroll
        for (int hB = 0; hB < 2; ++hB) {        // split B frags to cap VGPRs
            Frag16 bF[2];
#pragma unroll
            for (int q = 0; q < 2; ++q)
#pragma unroll
                for (int r = 0; r < 8; ++r)
                    bF[q].u[r] = *(const unsigned int*)
                        &Bt[(col0 + (hB * 2 + q) * 16 + lm) * LDT + kmap(r, hi)];
#pragma unroll
            for (int mt = 0; mt < 4; ++mt)
#pragma unroll
                for (int q = 0; q < 2; ++q)
                    acc[mt][hB * 2 + q].v = __builtin_amdgcn_wmma_f32_16x16x32_f16(
                        false, aF[mt].v, false, bF[q].v, (short)0,
                        acc[mt][hB * 2 + q].v, false, false);
        }
        __syncthreads();
    }

    // ---- Epilogue: C layout lane l -> N=l%16, VGPR r -> M=r+8*(l/16)
    const float wsc = (EPI == 2) ? wscalar[0] : 0.f;
#pragma unroll
    for (int mt = 0; mt < 4; ++mt) {
#pragma unroll
        for (int nt = 0; nt < 4; ++nt) {
            const int gn = n0 + col0 + nt * 16 + lm;
            const float bv = bias[gn];
#pragma unroll
            for (int r = 0; r < 8; ++r) {
                const int gm = m0 + row0 + mt * 16 + r + 8 * hi;
                float v = acc[mt][nt].f[r] + bv;
                const size_t oi = (size_t)gm * N + gn;
                if (EPI == 0) {
                    outh[oi] = (_Float16)v;
                } else if (EPI == 1) {
                    v = 0.5f * v * (1.0f + erff(v * 0.70710678118654752f));
                    outh[oi] = (_Float16)v;
                } else {
                    outf[oi] = res[oi] * wsc + v;
                }
            }
        }
    }
}

// ---------------------------------------------------------------------------
// Flash attention: one block per (b, h, 64-row q-tile); 128 thr = 4 waves,
// each wave owns 16 q rows. Loops 32 keys/iter. WMMA for QK^T and P*V.
// K tile copied via async-to-LDS; V transposed through VGPRs.
// qkv layout: [b, t, 3E] f16 (q at col h*64, k at 1024+h*64, v at 2048+h*64)
// ---------------------------------------------------------------------------
__global__ __launch_bounds__(128) void attn_kernel(const _Float16* __restrict__ qkv,
                                                   _Float16* __restrict__ y) {
    __shared__ __align__(16) _Float16 Ks[32 * 72];       // [key][dim]
    __shared__ __align__(16) _Float16 Vt[64 * 40];       // [dim][key]
    __shared__ __align__(16) _Float16 Ps[4 * 16 * 40];   // per-wave P tiles [m][k]

    const int tid  = threadIdx.x;
    const int lane = tid & 31;
    const int w    = tid >> 5;
    const int lm   = lane & 15;
    const int hi   = lane >> 4;

    const int qt = blockIdx.x;          // 0..31
    const int h  = blockIdx.y;          // 0..15
    const int b  = blockIdx.z;          // 0..1
    const int qbase = qt * 64;
    const int qrow0 = qbase + w * 16;

    // ---- Q fragments (16 rows x 64 dims = 2 A-frags), loaded once
    Frag16 qF[2];
#pragma unroll
    for (int t = 0; t < 2; ++t)
#pragma unroll
        for (int r = 0; r < 8; ++r)
            qF[t].u[r] = *(const unsigned int*)(qkv +
                (size_t)(b * TB + qrow0 + lm) * (3 * TE) + h * THD + t * 32 + kmap(r, hi));

    Acc8 accu[4];
    float mrow[8], lrow[8];
#pragma unroll
    for (int nt = 0; nt < 4; ++nt)
#pragma unroll
        for (int r = 0; r < 8; ++r) accu[nt].f[r] = 0.f;
#pragma unroll
    for (int r = 0; r < 8; ++r) { mrow[r] = -1e30f; lrow[r] = 0.f; }

    const int nk = qbase / 32 + 2;      // causal: keys up to qbase+63
    for (int it = 0; it < nk; ++it) {
        const int kb = it * 32;
        // ---- K tile [32 keys][64 dims] async; V tile transposed [64][32]
#pragma unroll
        for (int i = 0; i < 2; ++i) {
            const int idx = tid * 2 + i;          // 0..255
            const int key = idx >> 3;
            const int seg = idx & 7;
            const size_t base = (size_t)(b * TB + kb + key) * (3 * TE) + h * THD + seg * 8;
            copy_b128_to_lds(&Ks[key * 72 + seg * 8], qkv + base + TE);
            U4H vd;
            vd.q = *(const uint4*)(qkv + base + 2 * TE);
#pragma unroll
            for (int j = 0; j < 8; ++j) Vt[(seg * 8 + j) * 40 + key] = vd.h[j];
        }
        async_copies_wait();
        __syncthreads();

        // ---- S = Q @ K^T  (two 16x16 tiles over 32 keys, kdim=64 -> 2 wmma each)
        Acc8 S[2];
#pragma unroll
        for (int j = 0; j < 2; ++j) {
#pragma unroll
            for (int r = 0; r < 8; ++r) S[j].f[r] = 0.f;
#pragma unroll
            for (int t = 0; t < 2; ++t) {
                Frag16 kFr;
#pragma unroll
                for (int r = 0; r < 8; ++r)
                    kFr.u[r] = *(const unsigned int*)&Ks[(j * 16 + lm) * 72 + t * 32 + kmap(r, hi)];
                S[j].v = __builtin_amdgcn_wmma_f32_16x16x32_f16(
                    false, qF[t].v, false, kFr.v, (short)0, S[j].v, false, false);
            }
        }

        // ---- scale + causal mask + online softmax
        float sv[2][8];
#pragma unroll
        for (int j = 0; j < 2; ++j)
#pragma unroll
            for (int r = 0; r < 8; ++r) {
                const int key = kb + j * 16 + lm;
                const int qr  = qrow0 + r + 8 * hi;
                float s = S[j].f[r] * 0.125f;     // 1/sqrt(64)
                if (key > qr) s = -1e30f;
                sv[j][r] = s;
            }
        float alpha[8];
#pragma unroll
        for (int r = 0; r < 8; ++r) {
            const float rmax = warp16_max(fmaxf(sv[0][r], sv[1][r]));
            const float mnew = fmaxf(mrow[r], rmax);
            alpha[r] = __expf(mrow[r] - mnew);
            mrow[r]  = mnew;
            const float p0 = __expf(sv[0][r] - mnew);
            const float p1 = __expf(sv[1][r] - mnew);
            lrow[r] = lrow[r] * alpha[r] + warp16_sum(p0 + p1);
            // stash P (C-layout) into LDS for A-layout reload
            Ps[(w * 16 + r + 8 * hi) * 40 + 0 * 16 + lm] = (_Float16)p0;
            Ps[(w * 16 + r + 8 * hi) * 40 + 1 * 16 + lm] = (_Float16)p1;
        }
#pragma unroll
        for (int nt = 0; nt < 4; ++nt)
#pragma unroll
            for (int r = 0; r < 8; ++r) accu[nt].f[r] *= alpha[r];
        __syncthreads();

        // ---- O += P @ V   (A-frag from Ps, 4 B-frags from Vt)
        Frag16 pF;
#pragma unroll
        for (int r = 0; r < 8; ++r)
            pF.u[r] = *(const unsigned int*)&Ps[(w * 16 + lm) * 40 + kmap(r, hi)];
#pragma unroll
        for (int nt = 0; nt < 4; ++nt) {
            Frag16 vF;
#pragma unroll
            for (int r = 0; r < 8; ++r)
                vF.u[r] = *(const unsigned int*)&Vt[(nt * 16 + lm) * 40 + kmap(r, hi)];
            accu[nt].v = __builtin_amdgcn_wmma_f32_16x16x32_f16(
                false, pF.v, false, vF.v, (short)0, accu[nt].v, false, false);
        }
        __syncthreads();
    }

    // ---- normalize + store y[b, t, h*64+dim]
#pragma unroll
    for (int nt = 0; nt < 4; ++nt)
#pragma unroll
        for (int r = 0; r < 8; ++r) {
            const int row = qrow0 + r + 8 * hi;
            const int dim = nt * 16 + lm;
            y[(size_t)(b * TB + row) * TE + h * THD + dim] =
                (_Float16)(accu[nt].f[r] / lrow[r]);
        }
}

// ---------------------------------------------------------------------------
// Host launcher
// ---------------------------------------------------------------------------
extern "C" void kernel_launch(void* const* d_in, const int* in_sizes, int n_in,
                              void* d_out, int out_size, void* d_ws, size_t ws_size,
                              hipStream_t stream) {
    (void)in_sizes; (void)n_in; (void)out_size; (void)ws_size;

    const float* x      = (const float*)d_in[0];
    const float* ln1_g  = (const float*)d_in[1];
    const float* ln1_b  = (const float*)d_in[2];
    const float* attn_w = (const float*)d_in[3];
    const float* attn_b = (const float*)d_in[4];
    const float* proj_w = (const float*)d_in[5];
    const float* proj_b = (const float*)d_in[6];
    const float* ln2_g  = (const float*)d_in[7];
    const float* ln2_b  = (const float*)d_in[8];
    const float* mlp_w1 = (const float*)d_in[9];
    const float* mlp_b1 = (const float*)d_in[10];
    const float* mlp_w2 = (const float*)d_in[11];
    const float* mlp_b2 = (const float*)d_in[12];
    const float* w_att  = (const float*)d_in[13];
    const float* w_mlp  = (const float*)d_in[14];

    char* ws = (char*)d_ws;
    size_t off = 0;
    auto take = [&](size_t bytes) -> char* {
        char* p = ws + off;
        off += (bytes + 255) & ~(size_t)255;
        return p;
    };
    _Float16* xln   = (_Float16*)take((size_t)TM * TE * 2);
    _Float16* wqkvh = (_Float16*)take((size_t)TE * 3 * TE * 2);
    _Float16* qkv   = (_Float16*)take((size_t)TM * 3 * TE * 2);
    _Float16* ybuf  = (_Float16*)take((size_t)TM * TE * 2);
    _Float16* wprjh = (_Float16*)take((size_t)TE * TE * 2);
    float*    x1    = (float*)   take((size_t)TM * TE * 4);
    _Float16* x1ln  = (_Float16*)take((size_t)TM * TE * 2);
    _Float16* w1h   = (_Float16*)take((size_t)TE * 4 * TE * 2);
    _Float16* hbuf  = (_Float16*)take((size_t)TM * 4 * TE * 2);
    _Float16* w2h   = (_Float16*)take((size_t)4 * TE * TE * 2);

    // 1) LN1(x) -> f16
    ln_kernel<<<TM, 256, 0, stream>>>(x, ln1_g, ln1_b, xln);

    // 2) weights -> f16
    cast_kernel<<<1024, 256, 0, stream>>>(attn_w, wqkvh, TE * 3 * TE);
    cast_kernel<<<1024, 256, 0, stream>>>(proj_w, wprjh, TE * TE);
    cast_kernel<<<1024, 256, 0, stream>>>(mlp_w1, w1h, TE * 4 * TE);
    cast_kernel<<<1024, 256, 0, stream>>>(mlp_w2, w2h, 4 * TE * TE);

    // 3) QKV = LN1(x) @ attn_w + attn_b          [4096 x 3072]
    gemm_kernel<0><<<dim3(3 * TE / BN, TM / BM), 256, 0, stream>>>(
        xln, wqkvh, attn_b, qkv, nullptr, nullptr, nullptr, TM, 3 * TE, TE);

    // 4) attention -> y
    attn_kernel<<<dim3(TB / 64, TH, 2), 128, 0, stream>>>(qkv, ybuf);

    // 5) x1 = x*w_att + y @ proj_w + proj_b      [4096 x 1024]
    gemm_kernel<2><<<dim3(TE / BN, TM / BM), 256, 0, stream>>>(
        ybuf, wprjh, proj_b, nullptr, x1, x, w_att, TM, TE, TE);

    // 6) LN2(x1) -> f16
    ln_kernel<<<TM, 256, 0, stream>>>(x1, ln2_g, ln2_b, x1ln);

    // 7) h = gelu(LN2(x1) @ mlp_w1 + mlp_b1)     [4096 x 4096]
    gemm_kernel<1><<<dim3(4 * TE / BN, TM / BM), 256, 0, stream>>>(
        x1ln, w1h, mlp_b1, hbuf, nullptr, nullptr, nullptr, TM, 4 * TE, TE);

    // 8) out = x1*w_mlp + h @ mlp_w2 + mlp_b2    [4096 x 1024]
    gemm_kernel<2><<<dim3(TE / BN, TM / BM), 256, 0, stream>>>(
        hbuf, w2h, mlp_b2, nullptr, (float*)d_out, x1, w_mlp, TM, TE, 4 * TE);
}
